// MultiHeadAttention_3169685864466
// MI455X (gfx1250) — matrix operand order repaired and checked by reading, not compiled, not run on hardware
//
#include <hip/hip_runtime.h>
#include <stdint.h>

#define HIDDEN   1024
#define HEADS    16
#define HEAD_DIM 64
#define BATCH    4
#define SEQ      2048
#define MROWS    (BATCH * SEQ)   // 8192
#define N_QKV    (3 * HIDDEN)    // 3072

typedef __attribute__((ext_vector_type(16))) __bf16 v16bf;
typedef __attribute__((ext_vector_type(8)))  float  v8f;

// fp32 -> bf16 round-to-nearest-even
static __device__ __forceinline__ unsigned short f2bf(float f) {
  unsigned int u = __builtin_bit_cast(unsigned int, f);
  u += 0x7fffu + ((u >> 16) & 1u);
  return (unsigned short)(u >> 16);
}

// Load one 16-element bf16 fragment (32 bytes contiguous per lane).
static __device__ __forceinline__ v16bf load_frag(const void* p) {
  union { uint4 q[2]; v16bf v; } u;
  u.q[0] = *(const uint4*)p;
  u.q[1] = *((const uint4*)p + 1);
  return u.v;
}

static __device__ __forceinline__ v8f zero8() {
  v8f z = {0.f, 0.f, 0.f, 0.f, 0.f, 0.f, 0.f, 0.f};
  return z;
}

static __device__ __forceinline__ v8f wmma_bf16(v16bf a, v16bf b, v8f c) {
  return __builtin_amdgcn_wmma_f32_16x16x32_bf16(false, a, false, b, (short)0,
                                                 c, false, false);
}

// Async global -> LDS copy of 16 bytes per lane (CDNA5 ASYNCcnt path).
static __device__ __forceinline__ void async_copy_b128(unsigned lds_off,
                                                       const void* gaddr) {
  asm volatile("global_load_async_to_lds_b128 %0, %1, off"
               :: "v"(lds_off), "v"(gaddr)
               : "memory");
}
static __device__ __forceinline__ void wait_async0() {
  asm volatile("s_wait_asynccnt 0x0" ::: "memory");
}

// ---------------------------------------------------------------------------
// Pack fp32 row-major [Mrows, Kdim] into bf16 WMMA A-fragment layout:
//   [mt][kt][lane][e], lane&15 = m%16, lane>>4 = khalf,
//   K(e) = 32*kt + 8*khalf + (e&7) + ((e>>3)<<4)   (ISA 7.12.2, 16-bit A 16x32)
// ---------------------------------------------------------------------------
__global__ __launch_bounds__(256) void pack_a_f32_bf16(
    const float* __restrict__ src, unsigned short* __restrict__ dst,
    int mtiles, int ktiles, int ld) {
  int t = blockIdx.x * blockDim.x + threadIdx.x;
  if (t >= mtiles * ktiles * 32) return;
  int lane = t & 31;
  int kt   = (t >> 5) % ktiles;
  int mt   = (t >> 5) / ktiles;
  int m    = mt * 16 + (lane & 15);
  int kh   = lane >> 4;
  const float* row = src + (size_t)m * ld + kt * 32;
  __align__(16) unsigned short o[16];
#pragma unroll
  for (int e = 0; e < 16; ++e) {
    int K = 8 * kh + (e & 7) + ((e >> 3) << 4);
    o[e] = f2bf(row[K]);
  }
  unsigned short* p = dst + (size_t)t * 16;
  *(uint4*)p        = *(const uint4*)o;
  *(uint4*)(p + 8)  = *(const uint4*)(o + 8);
}

// ---------------------------------------------------------------------------
// Pack fp32 row-major [Kdim, N] into bf16 WMMA B-fragment layout:
//   [nt][kt][lane][e], lane&15 = n%16, lane>>4 = khalf, K(e) = 32*kt + 8*khalf + (e&7) + ((e>>3)<<4)
// ---------------------------------------------------------------------------
__global__ __launch_bounds__(256) void pack_b_f32_bf16(
    const float* __restrict__ src, unsigned short* __restrict__ dst,
    int ntiles, int ktiles, int ld) {
  int t = blockIdx.x * blockDim.x + threadIdx.x;
  if (t >= ntiles * ktiles * 32) return;
  int lane = t & 31;
  int kt   = (t >> 5) % ktiles;
  int nt   = (t >> 5) / ktiles;
  int n    = nt * 16 + (lane & 15);
  int kb   = kt * 32 + 8 * (lane >> 4);
  __align__(16) unsigned short o[16];
#pragma unroll
  for (int e = 0; e < 16; ++e) o[e] = f2bf(src[(size_t)(kb + (e & 7) + ((e >> 3) << 4)) * ld + n]);
  unsigned short* p = dst + (size_t)t * 16;
  *(uint4*)p        = *(const uint4*)o;
  *(uint4*)(p + 8)  = *(const uint4*)(o + 8);
}

// ---------------------------------------------------------------------------
// QKV projection: [8192,1024] x [1024,3072] + bias. Block = 8 waves sharing
// one 64-wide N strip. Each wave computes a 32x64 tile (2 A frags, 8 WMMAs /
// k-step). B tile (4 KB / k-step) double-buffered in LDS via
// global_load_async_to_lds_b128; A fragments software-pipelined across the
// per-step barrier so their global latency hides under the WMMAs.
// ---------------------------------------------------------------------------
__global__ __launch_bounds__(256) void qkv_gemm(
    const unsigned short* __restrict__ Xa, const unsigned short* __restrict__ Wp,
    const float* __restrict__ bias,
    unsigned short* __restrict__ Qp, unsigned short* __restrict__ Kp,
    unsigned short* __restrict__ Vp) {
  __shared__ __align__(16) unsigned char ldsB[2][4096];

  int tid  = threadIdx.x;
  int lane = tid & 31;
  int wave = blockIdx.x * 8 + (tid >> 5);
  int mt2 = wave % (MROWS / 32);     // 0..255 (pair of 16-row tiles)
  int nb  = wave / (MROWS / 32);     // 0..47  (64-wide N strip, same per block)
  int mta = mt2 * 2, mtb = mt2 * 2 + 1;

  // async staging role of this thread: fragment j0, 16B slice idx
  int j0  = tid >> 6;                // 0..3
  int idx = tid & 63;                // 0..63
  const char* wsrc =
      (const char*)Wp + (size_t)((nb * 4 + j0) * (HIDDEN / 32)) * 1024 + idx * 16;
  unsigned ldst[2];
  ldst[0] = (unsigned)(uintptr_t)&ldsB[0][j0 * 1024 + idx * 16];
  ldst[1] = (unsigned)(uintptr_t)&ldsB[1][j0 * 1024 + idx * 16];

  const unsigned short* apA = Xa + ((size_t)(mta * (HIDDEN / 32)) * 32 + lane) * 16;
  const unsigned short* apB = Xa + ((size_t)(mtb * (HIDDEN / 32)) * 32 + lane) * 16;

  v8f acc[2][4];
#pragma unroll
  for (int u = 0; u < 2; ++u)
#pragma unroll
    for (int j = 0; j < 4; ++j) acc[u][j] = zero8();

  auto step = [&](int p, v16bf a0, v16bf a1) {
    v16bf b0 = load_frag(&ldsB[p][0 * 1024 + lane * 32]);
    v16bf b1 = load_frag(&ldsB[p][1 * 1024 + lane * 32]);
    v16bf b2 = load_frag(&ldsB[p][2 * 1024 + lane * 32]);
    v16bf b3 = load_frag(&ldsB[p][3 * 1024 + lane * 32]);
    acc[0][0] = wmma_bf16(a0, b0, acc[0][0]);
    acc[1][0] = wmma_bf16(a1, b0, acc[1][0]);
    acc[0][1] = wmma_bf16(a0, b1, acc[0][1]);
    acc[1][1] = wmma_bf16(a1, b1, acc[1][1]);
    acc[0][2] = wmma_bf16(a0, b2, acc[0][2]);
    acc[1][2] = wmma_bf16(a1, b2, acc[1][2]);
    acc[0][3] = wmma_bf16(a0, b3, acc[0][3]);
    acc[1][3] = wmma_bf16(a1, b3, acc[1][3]);
  };

  async_copy_b128(ldst[0], wsrc);    // kt = 0 into buffer 0
  v16bf a0c = load_frag(apA);        // A fragments for kt = 0
  v16bf a1c = load_frag(apB);
#pragma unroll 1
  for (int kt = 0; kt < HIDDEN / 32 - 1; ++kt) {
    int p = kt & 1;
    wait_async0();                   // this wave's LDS writes done
    __syncthreads();                 // (waits DScnt) -> tile visible, prev reads done
    async_copy_b128(ldst[p ^ 1], wsrc + (size_t)(kt + 1) * 1024);
    // A fragments for kt+1 in flight under this step's WMMAs
    v16bf a0n = load_frag(apA + (size_t)(kt + 1) * 512);
    v16bf a1n = load_frag(apB + (size_t)(kt + 1) * 512);
    step(p, a0c, a1c);
    a0c = a0n;
    a1c = a1n;
  }
  wait_async0();
  __syncthreads();
  step((HIDDEN / 32 - 1) & 1, a0c, a1c);

  int hi = lane >> 4, nl = lane & 15;
#pragma unroll
  for (int u = 0; u < 2; ++u) {
#pragma unroll
    for (int j = 0; j < 4; ++j) {
#pragma unroll
      for (int r = 0; r < 8; ++r) {
        int m = (mt2 * 2 + u) * 16 + r + 8 * hi;
        int n = nb * 64 + j * 16 + nl;
        unsigned short bv = f2bf(acc[u][j][r] + bias[n]);
        int sel = n >> 10;          // 0=Q 1=K 2=V
        int h   = (n >> 6) & 15;
        int d   = n & 63;
        int bb  = m >> 11;
        int s   = m & 2047;
        size_t base = (size_t)(bb * HEADS + h) * (SEQ * HEAD_DIM);
        if (sel == 0) {             // Q: A-layout (m=s, kdim=d)
          int mq = s >> 4, ml = s & 15, kq = d >> 5, dd = d & 31;
          int e = (dd & 7) | ((dd >> 4) << 3), kh = (dd >> 3) & 1, l2 = kh * 16 + ml;
          Qp[base + ((size_t)(mq * 2 + kq) * 32 + l2) * 16 + e] = bv;
        } else if (sel == 1) {      // K: B-layout (n=s, kdim=d)  == K^T
          int nq = s >> 4, nl2 = s & 15, kq = d >> 5, dd = d & 31;
          int kh = (dd >> 3) & 1, e = (dd & 7) | ((dd >> 4) << 3), l2 = kh * 16 + nl2;
          Kp[base + ((size_t)(nq * 2 + kq) * 32 + l2) * 16 + e] = bv;
        } else {                    // V: B-layout (n=d, kdim=s)
          int nq = d >> 4, nl2 = d & 15, kq = s >> 5, ss = s & 31;
          int kh = (ss >> 3) & 1, e = (ss & 7) | ((ss >> 4) << 3), l2 = kh * 16 + nl2;
          Vp[base + ((size_t)(nq * (SEQ / 32) + kq) * 32 + l2) * 16 + e] = bv;
        }
      }
    }
  }
}

// ---------------------------------------------------------------------------
// Flash attention. Block = 8 waves sharing one (batch, head); per 32-key step
// the K chunk (4 KB contiguous) and V chunk (4x1 KB) are staged once per
// block into double-buffered LDS with global_load_async_to_lds_b128, then all
// waves read fragments via ds_load_b128 (8x less global K/V traffic).
// One wave owns 16 query rows x full head: 4 WMMA scores + online softmax +
// probs transpose through a per-wave LDS tile + 4 WMMA P*V.
// ---------------------------------------------------------------------------
__global__ __launch_bounds__(256) void attn_kernel(
    const unsigned short* __restrict__ Qp, const unsigned short* __restrict__ Kp,
    const unsigned short* __restrict__ Vp, unsigned short* __restrict__ Ap) {
  __shared__ __align__(16) unsigned char ldsK[2][4096];
  __shared__ __align__(16) unsigned char ldsV[2][4096];
  __shared__ __align__(16) unsigned short ldsP[8][16][32];

  int tid  = threadIdx.x;
  int lane = tid & 31;
  int w    = tid >> 5;
  int wave = blockIdx.x * 8 + w;
  int mt = wave & 127;               // query tile in sequence
  int h  = (wave >> 7) & 15;
  int b  = wave >> 11;
  int hi = lane >> 4, nl = lane & 15;

  size_t base = (size_t)(b * HEADS + h) * (SEQ * HEAD_DIM);
  const unsigned short* Qb = Qp + base;
  const char* kbytes = (const char*)(Kp + base);
  const char* vbytes = (const char*)(Vp + base);

  // staging roles
  int j0  = tid >> 6;                // 0..3
  int idx = tid & 63;                // 0..63
  unsigned kdst[2], vdst[2];
  kdst[0] = (unsigned)(uintptr_t)&ldsK[0][tid * 16];
  kdst[1] = (unsigned)(uintptr_t)&ldsK[1][tid * 16];
  vdst[0] = (unsigned)(uintptr_t)&ldsV[0][j0 * 1024 + idx * 16];
  vdst[1] = (unsigned)(uintptr_t)&ldsV[1][j0 * 1024 + idx * 16];

  auto issue_kv = [&](int p, int ck) {
    async_copy_b128(kdst[p], kbytes + (size_t)ck * 4096 + tid * 16);
    async_copy_b128(vdst[p], vbytes + (size_t)(j0 * (SEQ / 32) + ck) * 1024 + idx * 16);
  };

  v16bf qa0 = load_frag(Qb + ((size_t)(mt * 2 + 0) * 32 + lane) * 16);
  v16bf qa1 = load_frag(Qb + ((size_t)(mt * 2 + 1) * 32 + lane) * 16);

  v8f acc0 = zero8(), acc1 = zero8(), acc2 = zero8(), acc3 = zero8();
  float mi[8], li[8];
#pragma unroll
  for (int r = 0; r < 8; ++r) { mi[r] = -1e30f; li[r] = 0.f; }

  auto step = [&](int ck, int p) {
    // ---- scores: 16 queries x 32 keys (4 K frags preloaded) ---------------
    v16bf k00 = load_frag(&ldsK[p][0 * 1024 + lane * 32]);
    v16bf k01 = load_frag(&ldsK[p][1 * 1024 + lane * 32]);
    v16bf k10 = load_frag(&ldsK[p][2 * 1024 + lane * 32]);
    v16bf k11 = load_frag(&ldsK[p][3 * 1024 + lane * 32]);
    v8f sc[2];
    sc[0] = wmma_bf16(qa1, k01, wmma_bf16(qa0, k00, zero8()));
    sc[1] = wmma_bf16(qa1, k11, wmma_bf16(qa0, k10, zero8()));
    // ---- online softmax (row = r + 8*hi; 16 cols per lane-group) ----------
    float al[8];
#pragma unroll
    for (int r = 0; r < 8; ++r) {
      float s0 = sc[0][r] * 0.125f;   // 1/sqrt(64)
      float s1 = sc[1][r] * 0.125f;
      float pm = fmaxf(s0, s1);
#pragma unroll
      for (int o = 1; o < 16; o <<= 1) pm = fmaxf(pm, __shfl_xor(pm, o, 32));
      float nm = fmaxf(mi[r], pm);
      float a_ = __expf(mi[r] - nm);
      float p0 = __expf(s0 - nm);
      float p1 = __expf(s1 - nm);
      sc[0][r] = p0;
      sc[1][r] = p1;
      float ps = p0 + p1;
#pragma unroll
      for (int o = 1; o < 16; o <<= 1) ps += __shfl_xor(ps, o, 32);
      li[r] = li[r] * a_ + ps;
      mi[r] = nm;
      al[r] = a_;
    }
#pragma unroll
    for (int r = 0; r < 8; ++r) {
      acc0[r] *= al[r]; acc1[r] *= al[r]; acc2[r] *= al[r]; acc3[r] *= al[r];
    }
    // ---- probs C-frag -> bf16 A-frag through per-wave LDS tile ------------
#pragma unroll
    for (int t = 0; t < 2; ++t)
#pragma unroll
      for (int r = 0; r < 8; ++r)
        ldsP[w][r + 8 * hi][t * 16 + nl] = f2bf(sc[t][r]);

    const unsigned short* rp = &ldsP[w][nl][8 * hi];
    union { uint4 q[2]; v16bf v; } pa;
    pa.q[0] = *(const uint4*)rp;          // cols 8*hi .. 8*hi+7
    pa.q[1] = *(const uint4*)(rp + 16);   // cols 16+8*hi .. 16+8*hi+7
    // ---- P (16x32) @ V (32x64), 4 V frags preloaded -----------------------
    v16bf v0 = load_frag(&ldsV[p][0 * 1024 + lane * 32]);
    v16bf v1 = load_frag(&ldsV[p][1 * 1024 + lane * 32]);
    v16bf v2 = load_frag(&ldsV[p][2 * 1024 + lane * 32]);
    v16bf v3 = load_frag(&ldsV[p][3 * 1024 + lane * 32]);
    acc0 = wmma_bf16(pa.v, v0, acc0);
    acc1 = wmma_bf16(pa.v, v1, acc1);
    acc2 = wmma_bf16(pa.v, v2, acc2);
    acc3 = wmma_bf16(pa.v, v3, acc3);
  };

  issue_kv(0, 0);
#pragma unroll 1
  for (int ck = 0; ck < SEQ / 32 - 1; ++ck) {
    int p = ck & 1;
    wait_async0();
    __syncthreads();
    issue_kv(p ^ 1, ck + 1);
    step(ck, p);
  }
  wait_async0();
  __syncthreads();
  step(SEQ / 32 - 1, (SEQ / 32 - 1) & 1);

  // ---- normalize and store bf16 in A-layout for the output GEMM ----------
  v8f accs[4] = {acc0, acc1, acc2, acc3};
#pragma unroll
  for (int j = 0; j < 4; ++j) {
#pragma unroll
    for (int r = 0; r < 8; ++r) {
      float v = accs[j][r] / li[r];
      int s   = mt * 16 + r + 8 * hi;
      int row = b * SEQ + s;
      int K   = h * HEAD_DIM + j * 16 + nl;
      int mq = row >> 4, ml = row & 15, kq = K >> 5, dd = K & 31;
      int e = (dd & 7) | ((dd >> 4) << 3), kh = (dd >> 3) & 1, l2 = kh * 16 + ml;
      Ap[((size_t)(mq * (HIDDEN / 32) + kq) * 32 + l2) * 16 + e] = f2bf(v);
    }
  }
}

// ---------------------------------------------------------------------------
// Output projection: [8192,1024] x [1024,1024] + bias -> fp32 d_out.
// Same 32x64 wave tile + async-LDS B staging + pipelined A as qkv_gemm.
// ---------------------------------------------------------------------------
__global__ __launch_bounds__(256) void out_gemm(
    const unsigned short* __restrict__ Ap, const unsigned short* __restrict__ Wp,
    const float* __restrict__ bias, float* __restrict__ out) {
  __shared__ __align__(16) unsigned char ldsB[2][4096];

  int tid  = threadIdx.x;
  int lane = tid & 31;
  int wave = blockIdx.x * 8 + (tid >> 5);
  int mt2 = wave % (MROWS / 32);     // 0..255
  int nb  = wave / (MROWS / 32);     // 0..15
  int mta = mt2 * 2, mtb = mt2 * 2 + 1;

  int j0  = tid >> 6;
  int idx = tid & 63;
  const char* wsrc =
      (const char*)Wp + (size_t)((nb * 4 + j0) * (HIDDEN / 32)) * 1024 + idx * 16;
  unsigned ldst[2];
  ldst[0] = (unsigned)(uintptr_t)&ldsB[0][j0 * 1024 + idx * 16];
  ldst[1] = (unsigned)(uintptr_t)&ldsB[1][j0 * 1024 + idx * 16];

  const unsigned short* apA = Ap + ((size_t)(mta * (HIDDEN / 32)) * 32 + lane) * 16;
  const unsigned short* apB = Ap + ((size_t)(mtb * (HIDDEN / 32)) * 32 + lane) * 16;

  v8f acc[2][4];
#pragma unroll
  for (int u = 0; u < 2; ++u)
#pragma unroll
    for (int j = 0; j < 4; ++j) acc[u][j] = zero8();

  auto step = [&](int p, v16bf a0, v16bf a1) {
    v16bf b0 = load_frag(&ldsB[p][0 * 1024 + lane * 32]);
    v16bf b1 = load_frag(&ldsB[p][1 * 1024 + lane * 32]);
    v16bf b2 = load_frag(&ldsB[p][2 * 1024 + lane * 32]);
    v16bf b3 = load_frag(&ldsB[p][3 * 1024 + lane * 32]);
    acc[0][0] = wmma_bf16(a0, b0, acc[0][0]);
    acc[1][0] = wmma_bf16(a1, b0, acc[1][0]);
    acc[0][1] = wmma_bf16(a0, b1, acc[0][1]);
    acc[1][1] = wmma_bf16(a1, b1, acc[1][1]);
    acc[0][2] = wmma_bf16(a0, b2, acc[0][2]);
    acc[1][2] = wmma_bf16(a1, b2, acc[1][2]);
    acc[0][3] = wmma_bf16(a0, b3, acc[0][3]);
    acc[1][3] = wmma_bf16(a1, b3, acc[1][3]);
  };

  async_copy_b128(ldst[0], wsrc);
  v16bf a0c = load_frag(apA);
  v16bf a1c = load_frag(apB);
#pragma unroll 1
  for (int kt = 0; kt < HIDDEN / 32 - 1; ++kt) {
    int p = kt & 1;
    wait_async0();
    __syncthreads();
    async_copy_b128(ldst[p ^ 1], wsrc + (size_t)(kt + 1) * 1024);
    v16bf a0n = load_frag(apA + (size_t)(kt + 1) * 512);
    v16bf a1n = load_frag(apB + (size_t)(kt + 1) * 512);
    step(p, a0c, a1c);
    a0c = a0n;
    a1c = a1n;
  }
  wait_async0();
  __syncthreads();
  step((HIDDEN / 32 - 1) & 1, a0c, a1c);

  int hi = lane >> 4, nl = lane & 15;
#pragma unroll
  for (int u = 0; u < 2; ++u) {
#pragma unroll
    for (int j = 0; j < 4; ++j) {
#pragma unroll
      for (int r = 0; r < 8; ++r) {
        int m = (mt2 * 2 + u) * 16 + r + 8 * hi;
        int n = nb * 64 + j * 16 + nl;
        out[(size_t)m * HIDDEN + n] = acc[u][j][r] + bias[n];
      }
    }
  }
}

// ---------------------------------------------------------------------------
extern "C" void kernel_launch(void* const* d_in, const int* in_sizes, int n_in,
                              void* d_out, int out_size, void* d_ws, size_t ws_size,
                              hipStream_t stream) {
  (void)in_sizes; (void)n_in; (void)out_size; (void)ws_size;
  const float* x    = (const float*)d_in[0];
  const float* Wqkv = (const float*)d_in[1];
  const float* bqkv = (const float*)d_in[2];
  const float* Wout = (const float*)d_in[3];
  const float* bout = (const float*)d_in[4];
  float* out = (float*)d_out;

  char* ws = (char*)d_ws;
  size_t off = 0;
  auto take = [&](size_t bytes) -> char* {
    char* p = ws + off;
    off += (bytes + 255) & ~(size_t)255;
    return p;
  };
  unsigned short* Xa  = (unsigned short*)take((size_t)MROWS * HIDDEN * 2);   // 16 MB
  unsigned short* Wqp = (unsigned short*)take((size_t)HIDDEN * N_QKV * 2);   // 6 MB
  unsigned short* Wop = (unsigned short*)take((size_t)HIDDEN * HIDDEN * 2);  // 2 MB
  unsigned short* Qp  = (unsigned short*)take((size_t)MROWS * HIDDEN * 2);   // 16 MB
  unsigned short* Kp  = (unsigned short*)take((size_t)MROWS * HIDDEN * 2);   // 16 MB
  unsigned short* Vp  = (unsigned short*)take((size_t)MROWS * HIDDEN * 2);   // 16 MB
  unsigned short* Ap  = (unsigned short*)take((size_t)MROWS * HIDDEN * 2);   // 16 MB

  // 1) pack operands into WMMA-native fragment layouts (fp32 -> bf16)
  pack_a_f32_bf16<<<(MROWS / 16) * (HIDDEN / 32) * 32 / 256, 256, 0, stream>>>(
      x, Xa, MROWS / 16, HIDDEN / 32, HIDDEN);
  pack_b_f32_bf16<<<(N_QKV / 16) * (HIDDEN / 32) * 32 / 256, 256, 0, stream>>>(
      Wqkv, Wqp, N_QKV / 16, HIDDEN / 32, N_QKV);
  pack_b_f32_bf16<<<(HIDDEN / 16) * (HIDDEN / 32) * 32 / 256, 256, 0, stream>>>(
      Wout, Wop, HIDDEN / 16, HIDDEN / 32, HIDDEN);

  // 2) QKV projection (WMMA, 32x64 wave tiles, async-LDS B staging)
  qkv_gemm<<<(MROWS / 32) * (N_QKV / 64) / 8, 256, 0, stream>>>(Xa, Wqp, bqkv, Qp, Kp, Vp);

  // 3) flash attention (WMMA + async-LDS K/V staging, online softmax)
  attn_kernel<<<BATCH * HEADS * (SEQ / 16) / 8, 256, 0, stream>>>(Qp, Kp, Vp, Ap);

  // 4) output projection (WMMA, 32x64 wave tiles, async-LDS B staging)
  out_gemm<<<(MROWS / 32) * (HIDDEN / 64) / 8, 256, 0, stream>>>(Ap, Wop, bout, out);
}
